// LocalGridAlignAttention_7000796692589
// MI455X (gfx1250) — compile-verified, matrix operand order checked by reading, not run on hardware
//
#include <hip/hip_runtime.h>
#include <hip/hip_bf16.h>

typedef __attribute__((ext_vector_type(16))) __bf16 v16bf;
typedef __attribute__((ext_vector_type(8)))  __bf16 v8bf;
typedef __attribute__((ext_vector_type(8)))  float  v8f;

#define CCH   256          // channels
#define HD    32           // head dim
#define TQ    16           // query pixels per workgroup
#define VSTR  264          // LDS row stride (bf16 elems), 16B-aligned rows, bank-skewed
#define QSTR  256          // sQ row stride (f32 elems)
#define SCALE 0.17677669529663687f   // 1/sqrt(32)

// LDS layout (bytes): sV | sQ | (sX overlaid with sE)
#define SV_BYTES  (256 * VSTR * 2)          // 135168
#define SQ_OFF    SV_BYTES
#define SQ_BYTES  (TQ * QSTR * 4)           // 16384
#define SXE_OFF   (SQ_OFF + SQ_BYTES)       // 151552
#define SXE_BYTES (TQ * VSTR * 2)           // 8448 (>= 128*16*4 = 8192 for sE)
#define SMEM_TOTAL (SXE_OFF + SXE_BYTES)    // 160000

__device__ __forceinline__ v16bf ld16(const __bf16* p) {
  // 16-bit A/B fragment: K = kb..kb+7 at p, K = kb+16..kb+23 at p+16 elems
  union { v16bf v; v8bf h[2]; } u;
  u.h[0] = *(const v8bf*)(p);
  u.h[1] = *(const v8bf*)(p + 16);
  return u.v;
}

__global__ void cvt_bf16_kernel(const float* __restrict__ in, __bf16* __restrict__ o, int n) {
  int i = blockIdx.x * blockDim.x + threadIdx.x;
  if (i < n) o[i] = (__bf16)in[i];
}

__global__ void __launch_bounds__(256)
lga_main(const float* __restrict__ x, const float* __restrict__ featmap,
         const __bf16* __restrict__ wq_bf, const float* __restrict__ bq,
         const __bf16* __restrict__ wk_bf, const float* __restrict__ bk,
         float* __restrict__ out)
{
  extern __shared__ char smem[];
  __bf16* sV = (__bf16*)(smem);             // [256 win px][VSTR] bf16
  float*  sQ = (float*)(smem + SQ_OFF);     // [16 qpix][QSTR] f32 (scaled, biased)
  __bf16* sX = (__bf16*)(smem + SXE_OFF);   // [16 qpix][VSTR] bf16 (phase 1 only)
  float*  sE = (float*)(smem + SXE_OFF);    // [16 qpix][8 heads][16 off] f32 (after)

  const int t    = threadIdx.x;
  const int wave = t >> 5, lane = t & 31;
  const int half = lane >> 4, l16 = lane & 15;

  const int bid  = blockIdx.x;              // 2048 = B(8) * H(64) * Wblk(4)
  const int wblk = bid & 3;
  const int h    = (bid >> 2) & 63;
  const int b    = bid >> 8;
  const int w0   = wblk * TQ;

  // ---- phase 0a: stage x tile (16 px x 256 ch) as bf16, float4 loads ----
  {
    const int qg = (t & 3) * 4;             // 4-pixel group along W
    for (int c = t >> 2; c < CCH; c += 64) {
      const float4 v = *(const float4*)(x + (((size_t)b * CCH + c) * 64 + h) * 64 + w0 + qg);
      sX[(qg + 0) * VSTR + c] = (__bf16)v.x;
      sX[(qg + 1) * VSTR + c] = (__bf16)v.y;
      sX[(qg + 2) * VSTR + c] = (__bf16)v.z;
      sX[(qg + 3) * VSTR + c] = (__bf16)v.w;
    }
  }
  // ---- phase 0b: stage featmap window (4 rows x 64 cols x 256 ch), float4 loads ----
  {
    const int cg  = (t & 15) * 4;           // 4-col group
    const int c0  = (t >> 4) & 3;
    const int p   = t >> 6;                 // window row
    const size_t fbase = (((size_t)b * CCH) * 256 + (4 * h + p)) * 256 + (4 * w0 + cg);
    for (int c = c0; c < CCH; c += 4) {
      const float4 v = *(const float4*)(featmap + fbase + (size_t)c * 65536);
      sV[(p * 64 + cg + 0) * VSTR + c] = (__bf16)v.x;
      sV[(p * 64 + cg + 1) * VSTR + c] = (__bf16)v.y;
      sV[(p * 64 + cg + 2) * VSTR + c] = (__bf16)v.z;
      sV[(p * 64 + cg + 3) * VSTR + c] = (__bf16)v.w;
    }
  }
  __syncthreads();

  // ---- phase 1: Q = (X * Wq^T + bq) * scale via WMMA bf16 (A hoisted) ----
  {
    v16bf aF[8];
    #pragma unroll
    for (int kk = 0; kk < 8; ++kk)
      aF[kk] = ld16(&sX[l16 * VSTR + kk * 32 + half * 8]);   // A: M=qpix, K=chan
    for (int nt = wave; nt < 16; nt += 8) {
      const int n0 = nt * 16;
      v8f acc = {0.f, 0.f, 0.f, 0.f, 0.f, 0.f, 0.f, 0.f};
      #pragma unroll
      for (int kk = 0; kk < 8; ++kk) {
        v16bf bb = ld16(&wq_bf[(size_t)(n0 + l16) * CCH + kk * 32 + half * 8]);
        acc = __builtin_amdgcn_wmma_f32_16x16x32_bf16(
            false, aF[kk], false, bb, (short)0, acc, false, false);
      }
      const float bias = bq[n0 + l16];
      #pragma unroll
      for (int r = 0; r < 8; ++r)   // D: lane holds N=n0+l16, M=half*8+r
        sQ[(half * 8 + r) * QSTR + n0 + l16] = (acc[r] + bias) * SCALE;
    }
  }
  __syncthreads();

  // ---- energy init: e0[qp][head] = dot(Qscaled[qp, head], bk[head]) ----
  if (t < 128) {
    const int qp = t >> 3, hm = t & 7;
    float e0 = 0.f;
    for (int cl = 0; cl < HD; ++cl)
      e0 += sQ[qp * QSTR + hm * HD + cl] * bk[hm * HD + cl];
    #pragma unroll
    for (int j = 0; j < 16; ++j) sE[(qp * 8 + hm) * 16 + j] = e0;
  }
  __syncthreads();

  // ---- phase 2: K projection via WMMA + on-the-fly energy accumulation ----
  // Each wave owns 2 m-tiles; A fragments for all 8 K-steps held in registers,
  // only B (weights, L2-resident) is re-streamed per n-tile.
  for (int mt = wave; mt < 16; mt += 8) {
    const int wp0 = mt * 16;
    v16bf aF[8];
    #pragma unroll
    for (int kk = 0; kk < 8; ++kk)
      aF[kk] = ld16(&sV[(wp0 + l16) * VSTR + kk * 32 + half * 8]);  // A: M=win px
    for (int nt = 0; nt < 16; ++nt) {
      const int n0 = nt * 16;
      const int hm = n0 >> 5;               // head of this 16-channel slab
      v8f acc = {0.f, 0.f, 0.f, 0.f, 0.f, 0.f, 0.f, 0.f};
      #pragma unroll
      for (int kk = 0; kk < 8; ++kk) {
        v16bf bb = ld16(&wk_bf[(size_t)(n0 + l16) * CCH + kk * 32 + half * 8]);
        acc = __builtin_amdgcn_wmma_f32_16x16x32_bf16(
            false, aF[kk], false, bb, (short)0, acc, false, false);
      }
      // each D row r: one window pixel, 16 channels striped across the 16-lane half
      #pragma unroll
      for (int r = 0; r < 8; ++r) {
        const int wp  = wp0 + half * 8 + r;      // window pixel = p*64 + col
        const int col = wp & 63;
        const int qp  = col >> 2;                // owning query pixel
        const int off = ((wp >> 6) << 2) | (col & 3);   // window offset p*4+q
        float v = acc[r] * sQ[qp * QSTR + n0 + l16];
        v += __shfl_xor(v, 1, 32);
        v += __shfl_xor(v, 2, 32);
        v += __shfl_xor(v, 4, 32);
        v += __shfl_xor(v, 8, 32);               // reduce 16 channels within half
        if (l16 == 0) atomicAdd(&sE[(qp * 8 + hm) * 16 + off], v);
      }
    }
  }
  __syncthreads();

  // ---- phase 3: softmax over the 16 window positions, per (qpix, head) ----
  if (t < 128) {
    const int base = t * 16;
    float mx = sE[base];
    #pragma unroll
    for (int j = 1; j < 16; ++j) mx = fmaxf(mx, sE[base + j]);
    float s = 0.f;
    #pragma unroll
    for (int j = 0; j < 16; ++j) { float e = __expf(sE[base + j] - mx); sE[base + j] = e; s += e; }
    const float inv = 1.f / s;
    #pragma unroll
    for (int j = 0; j < 16; ++j) sE[base + j] *= inv;
  }
  __syncthreads();

  // ---- phase 4: aggregate V (sparse 16-position gather) + write out ----
  {
    const int qp = t & 15, cb = t >> 4;
    #pragma unroll
    for (int i = 0; i < 16; ++i) {
      const int ch = cb * 16 + i;
      const float* att = &sE[(qp * 8 + (ch >> 5)) * 16];
      float o = 0.f;
      #pragma unroll
      for (int pq = 0; pq < 16; ++pq) {
        const int p = pq >> 2, qf = pq & 3;
        o += att[pq] * (float)sV[(p * 64 + qp * 4 + qf) * VSTR + ch];
      }
      out[(((size_t)b * CCH + ch) * 64 + h) * 64 + w0 + qp] = o;
    }
  }
}

extern "C" void kernel_launch(void* const* d_in, const int* in_sizes, int n_in,
                              void* d_out, int out_size, void* d_ws, size_t ws_size,
                              hipStream_t stream) {
  const float* x  = (const float*)d_in[0];
  const float* fm = (const float*)d_in[1];
  const float* wq = (const float*)d_in[2];
  const float* bq = (const float*)d_in[3];
  const float* wk = (const float*)d_in[4];
  const float* bk = (const float*)d_in[5];
  float* out = (float*)d_out;

  __bf16* wq_bf = (__bf16*)d_ws;
  __bf16* wk_bf = (__bf16*)((char*)d_ws + 65536 * 2);

  cvt_bf16_kernel<<<256, 256, 0, stream>>>(wq, wq_bf, 65536);
  cvt_bf16_kernel<<<256, 256, 0, stream>>>(wk, wk_bf, 65536);

  // grid: B(8) * H(64) * 4 w-blocks = 2048 workgroups of 256 threads (8 waves)
  lga_main<<<2048, 256, SMEM_TOTAL, stream>>>(x, fm, wq_bf, bq, wk_bf, bk, out);
}
